// GraphNetwork_41188736369264
// MI455X (gfx1250) — matrix operand order repaired and checked by reading, not compiled, not run on hardware
//
#include <hip/hip_runtime.h>
#include <hip/hip_bf16.h>

// ---------------------------------------------------------------------------
// Types for CDNA5 WMMA (wave32): v_wmma_f32_16x16x32_bf16
// ---------------------------------------------------------------------------
typedef float  v8f   __attribute__((ext_vector_type(8)));
typedef __bf16 v16bf __attribute__((ext_vector_type(16)));
typedef __bf16 v8bf  __attribute__((ext_vector_type(8)));

// ---------------------------------------------------------------------------
// A-fragment loaders (16x32 bf16 tile, CDNA5 layout):
//   lane l<16 : row=l,    halves 0..7 -> K=k1..k1+7,  halves 8..15 -> K=k1+16..k1+23
//   lane l>=16: row=l-16, k1 includes +8 (kbase)
// Four explicit float4 temporaries -> loads form one clause before converts.
// ---------------------------------------------------------------------------
__device__ inline v16bf load_afrag(const float* __restrict__ p, int k1) {
  float4 f0 = *(const float4*)(p + k1);
  float4 f1 = *(const float4*)(p + k1 + 4);
  float4 f2 = *(const float4*)(p + k1 + 16);
  float4 f3 = *(const float4*)(p + k1 + 20);
  v16bf a;
  a[0]  = (__bf16)f0.x; a[1]  = (__bf16)f0.y; a[2]  = (__bf16)f0.z; a[3]  = (__bf16)f0.w;
  a[4]  = (__bf16)f1.x; a[5]  = (__bf16)f1.y; a[6]  = (__bf16)f1.z; a[7]  = (__bf16)f1.w;
  a[8]  = (__bf16)f2.x; a[9]  = (__bf16)f2.y; a[10] = (__bf16)f2.z; a[11] = (__bf16)f2.w;
  a[12] = (__bf16)f3.x; a[13] = (__bf16)f3.y; a[14] = (__bf16)f3.z; a[15] = (__bf16)f3.w;
  return a;
}

__device__ inline v16bf load_afrag(const __bf16* __restrict__ p, int k1) {
  v8bf lo = *(const v8bf*)(p + k1);
  v8bf hi = *(const v8bf*)(p + k1 + 16);
  return __builtin_shufflevector(lo, hi, 0,1,2,3,4,5,6,7,8,9,10,11,12,13,14,15);
}

// ---------------------------------------------------------------------------
// GEMM: C[M,N](bf16) = A[M,K] x B[K,N], B given transposed bf16 (BT[N][K]).
// One wave per 16-row x 128-col strip: 8 C-fragments.
// Per K-step: all 8 B fragments loaded in one clause, next A prefetched,
// then 8 WMMAs — overlaps vmem latency with the matrix pipe.
// grid = (M/16, N/128), block = 32 (one wave).
// ---------------------------------------------------------------------------
template <typename AT>
__global__ __launch_bounds__(32)
void gemm_bf16_wmma(const AT* __restrict__ A, int lda,
                    const __bf16* __restrict__ BT,
                    __bf16* __restrict__ Cout, int ldc, int K) {
  const int lane = threadIdx.x;
  const int hsel = lane >> 4;      // 0: K-low half lanes, 1: K-high half lanes
  const int l15  = lane & 15;
  const long row = (long)blockIdx.x * 16 + l15;
  const int  colBase = blockIdx.y * 128;
  const AT* Arow = A + row * (long)lda;
  const int kb = hsel * 8;

  v8f acc[8] = {};

  v16bf a = load_afrag(Arow, kb);            // K-step 0 A fragment

  for (int k0 = 0; k0 < K; k0 += 32) {
    // B 32x16 frags: lanes 0-15 hold K=k0..k0+7,k0+16..k0+23 for col=l15; lanes 16-31 K=k0+8..k0+15,k0+24..k0+31
    v16bf bf[8];
#pragma unroll
    for (int t = 0; t < 8; ++t) {
      const __bf16* bp = BT + (long)(colBase + t * 16 + l15) * K + k0 + hsel * 8;
      v8bf b0 = *(const v8bf*)bp;
      v8bf b1 = *(const v8bf*)(bp + 16);
      bf[t] = __builtin_shufflevector(b0, b1, 0,1,2,3,4,5,6,7,8,9,10,11,12,13,14,15);
    }
    // prefetch next K-step's A while this step's WMMAs run
    v16bf anext = a;
    if (k0 + 32 < K) anext = load_afrag(Arow, k0 + 32 + kb);
#pragma unroll
    for (int t = 0; t < 8; ++t) {
      acc[t] = __builtin_amdgcn_wmma_f32_16x16x32_bf16(
          false, a, false, bf[t], (short)0, acc[t], false, false);
    }
    a = anext;
  }

  // C 16x16 f32 frag: vgpr v -> row = v + 8*hsel, col = l15
  const long rbase = (long)blockIdx.x * 16 + hsel * 8;
#pragma unroll
  for (int t = 0; t < 8; ++t) {
    const int col = colBase + t * 16 + l15;
#pragma unroll
    for (int v = 0; v < 8; ++v)
      Cout[(rbase + v) * (long)ldc + col] = (__bf16)acc[t][v];
  }
}

// ---------------------------------------------------------------------------
// Build transposed bf16 fused weight matrix WT[nblk*O][K]:
//   block b -> relation rel_b >= 0: W = coeff[rel,0]*basis0 + coeff[rel,1]*basis1
//           -> rel_b == -1       : W = loop_w
// ---------------------------------------------------------------------------
__global__ void build_wt(const float* __restrict__ coeff,
                         const float* __restrict__ basis,
                         const float* __restrict__ loopw,
                         __bf16* __restrict__ out,
                         int K, int O, int r0, int r1, int r2, int nblk) {
  long idx = (long)blockIdx.x * blockDim.x + threadIdx.x;
  long total = (long)nblk * O * K;
  if (idx >= total) return;
  int k = (int)(idx % K);
  int n = (int)(idx / K);
  int blk = n / O, o = n % O;
  int rel = (blk == 0) ? r0 : ((blk == 1) ? r1 : r2);
  float v;
  if (rel < 0) {
    v = loopw[(long)k * O + o];
  } else {
    v = coeff[rel * 2 + 0] * basis[(long)k * O + o] +
        coeff[rel * 2 + 1] * basis[(long)K * O + (long)k * O + o];
  }
  out[(long)n * K + k] = (__bf16)v;
}

// ---------------------------------------------------------------------------
// In-degree (float) per destination node
// ---------------------------------------------------------------------------
__global__ void degree_k(const int* __restrict__ dst, int n, float* __restrict__ deg) {
  int i = blockIdx.x * blockDim.x + threadIdx.x;
  if (i < n) atomicAdd(&deg[dst[i]], 1.0f);
}

// ---------------------------------------------------------------------------
// Edge scatter: agg[dst] += Y[src, coloff:coloff+F]; thread = (edge, 4 feats)
// ---------------------------------------------------------------------------
__global__ void scatter_add(const __bf16* __restrict__ Y, int ldy, int coloff,
                            const int* __restrict__ src, const int* __restrict__ dst,
                            int nedges, float* __restrict__ agg, int F) {
  long tid = (long)blockIdx.x * blockDim.x + threadIdx.x;
  int fv = F >> 2;
  long total = (long)nedges * fv;
  if (tid >= total) return;
  int e = (int)(tid / fv);
  int j = (int)(tid % fv) * 4;
  int s = src[e], d = dst[e];
  const __bf16* yp = Y + (long)s * ldy + coloff + j;
  float* ap = agg + (long)d * F + j;
  atomicAdd(ap + 0, (float)yp[0]);
  atomicAdd(ap + 1, (float)yp[1]);
  atomicAdd(ap + 2, (float)yp[2]);
  atomicAdd(ap + 3, (float)yp[3]);
}

// ---------------------------------------------------------------------------
// Layer-1 combine: h = relu(aggA/deg + [aggB/deg] + loop + bias) -> bf16
// ---------------------------------------------------------------------------
__global__ void combine1(const float* __restrict__ aggA, const float* __restrict__ degA,
                         const float* __restrict__ aggB, const float* __restrict__ degB,
                         const __bf16* __restrict__ Yloop, int ldy, int coloff,
                         const float* __restrict__ bias,
                         __bf16* __restrict__ hout, int F, int n) {
  long idx = (long)blockIdx.x * blockDim.x + threadIdx.x;
  if (idx >= (long)n * F) return;
  int i = (int)(idx / F), f = (int)(idx % F);
  float v = aggA[(long)i * F + f] / fmaxf(degA[i], 1.0f);
  if (aggB) v += aggB[(long)i * F + f] / fmaxf(degB[i], 1.0f);
  v += (float)Yloop[(long)i * ldy + coloff + f] + bias[f];
  hout[(long)i * F + f] = (__bf16)fmaxf(v, 0.0f);
}

// ---------------------------------------------------------------------------
// Layer-2 combine fused with readout: one wave per node, F=128 fixed.
// score = relu(h2) . w_score, wave-reduced via shfl, atomic into out[gid].
// ---------------------------------------------------------------------------
__global__ __launch_bounds__(256)
void combine2_readout(const float* __restrict__ aggA, const float* __restrict__ degA,
                      const float* __restrict__ aggB, const float* __restrict__ degB,
                      const __bf16* __restrict__ Yloop, int ldy, int coloff,
                      const float* __restrict__ bias, const float* __restrict__ wscore,
                      const int* __restrict__ gid, float* __restrict__ out, int n) {
  int lane = threadIdx.x & 31;
  int node = blockIdx.x * 8 + (threadIdx.x >> 5);
  if (node >= n) return;                     // uniform per wave
  float dA = fmaxf(degA[node], 1.0f);
  float dB = aggB ? fmaxf(degB[node], 1.0f) : 1.0f;
  const float*  pA = aggA + (long)node * 128;
  const float*  pB = aggB ? (aggB + (long)node * 128) : nullptr;
  const __bf16* pL = Yloop + (long)node * ldy + coloff;
  float partial = 0.0f;
  int f0 = lane * 4;
#pragma unroll
  for (int j = 0; j < 4; ++j) {
    int f = f0 + j;
    float v = pA[f] / dA;
    if (pB) v += pB[f] / dB;
    v += (float)pL[f] + bias[f];
    v = fmaxf(v, 0.0f);
    partial += v * wscore[f];
  }
#pragma unroll
  for (int off = 16; off >= 1; off >>= 1)
    partial += __shfl_xor(partial, off, 32);
  if (lane == 0) atomicAdd(&out[gid[node]], partial);
}

__global__ void out_init(float* __restrict__ out, const float* __restrict__ b, int g) {
  int i = threadIdx.x;
  if (i < g) out[i] = b[0];
}

// ---------------------------------------------------------------------------
static inline size_t alignup(size_t x) { return (x + 255) & ~(size_t)255; }

extern "C" void kernel_launch(void* const* d_in, const int* in_sizes, int n_in,
                              void* d_out, int out_size, void* d_ws, size_t ws_size,
                              hipStream_t stream) {
  const float* x_sent  = (const float*)d_in[0];
  const float* x_doc   = (const float*)d_in[1];
  const float* coeff1  = (const float*)d_in[2];
  const float* basis1  = (const float*)d_in[3];
  const float* loop_w1 = (const float*)d_in[4];
  const float* bias1   = (const float*)d_in[5];
  const float* coeff2  = (const float*)d_in[6];
  const float* basis2  = (const float*)d_in[7];
  const float* loop_w2 = (const float*)d_in[8];
  const float* bias2   = (const float*)d_in[9];
  const float* w_score = (const float*)d_in[10];
  const float* b_score = (const float*)d_in[11];
  const int* src_ss = (const int*)d_in[12];
  const int* dst_ss = (const int*)d_in[13];
  const int* src_sd = (const int*)d_in[14];
  const int* dst_sd = (const int*)d_in[15];
  const int* src_ds = (const int*)d_in[16];
  const int* dst_ds = (const int*)d_in[17];
  const int* gid_sent = (const int*)d_in[18];
  const int* gid_doc  = (const int*)d_in[19];

  const int NS = in_sizes[0] / 768;   // 100000
  const int ND = in_sizes[1] / 768;   // 10000
  const int Ess = in_sizes[12], Esd = in_sizes[14], Eds = in_sizes[16];
  const int G = out_size;             // 32
  float* out = (float*)d_out;

  // ---- workspace carve-out -------------------------------------------------
  char* ws = (char*)d_ws;
  size_t off = 0;
  auto carve = [&](size_t bytes) -> void* { void* p = ws + off; off = alignup(off + bytes); return p; };
  float*  degSs = (float*)carve((size_t)NS * 4);
  float*  degDs = (float*)carve((size_t)NS * 4);
  float*  degSd = (float*)carve((size_t)ND * 4);
  __bf16* WTs1  = (__bf16*)carve((size_t)768 * 768 * 2);
  __bf16* WTd1  = (__bf16*)carve((size_t)512 * 768 * 2);
  __bf16* WTs2  = (__bf16*)carve((size_t)384 * 256 * 2);
  __bf16* WTd2  = (__bf16*)carve((size_t)256 * 256 * 2);
  __bf16* Ys    = (__bf16*)carve((size_t)NS * 768 * 2);   // layer1 sent GEMM out; reused as Y2s [NS,384]
  __bf16* Yd    = (__bf16*)carve((size_t)ND * 512 * 2);   // layer1 doc GEMM out; reused as Y2d [ND,256]
  float*  aggA  = (float*)carve((size_t)NS * 256 * 4);    // sent <- ds
  float*  aggB  = (float*)carve((size_t)NS * 256 * 4);    // sent <- ss
  float*  aggD  = (float*)carve((size_t)ND * 256 * 4);    // doc  <- sd
  __bf16* h1s   = (__bf16*)carve((size_t)NS * 256 * 2);
  __bf16* h1d   = (__bf16*)carve((size_t)ND * 256 * 2);
  (void)ws_size; (void)n_in;

  // ---- degrees (once per launch, reused by both layers) --------------------
  hipMemsetAsync(degSs, 0, (size_t)NS * 4, stream);
  hipMemsetAsync(degDs, 0, (size_t)NS * 4, stream);
  hipMemsetAsync(degSd, 0, (size_t)ND * 4, stream);
  degree_k<<<(Ess + 255) / 256, 256, 0, stream>>>(dst_ss, Ess, degSs);
  degree_k<<<(Eds + 255) / 256, 256, 0, stream>>>(dst_ds, Eds, degDs);
  degree_k<<<(Esd + 255) / 256, 256, 0, stream>>>(dst_sd, Esd, degSd);

  // ---- fused transposed weights (bf16) -------------------------------------
  {
    long t1 = (long)3 * 256 * 768;
    build_wt<<<(t1 + 255) / 256, 256, 0, stream>>>(coeff1, basis1, loop_w1, WTs1, 768, 256, 0, 2, -1, 3);
    long t2 = (long)2 * 256 * 768;
    build_wt<<<(t2 + 255) / 256, 256, 0, stream>>>(coeff1, basis1, loop_w1, WTd1, 768, 256, 1, -1, -1, 2);
    long t3 = (long)3 * 128 * 256;
    build_wt<<<(t3 + 255) / 256, 256, 0, stream>>>(coeff2, basis2, loop_w2, WTs2, 256, 128, 0, 2, -1, 3);
    long t4 = (long)2 * 128 * 256;
    build_wt<<<(t4 + 255) / 256, 256, 0, stream>>>(coeff2, basis2, loop_w2, WTd2, 256, 128, 1, -1, -1, 2);
  }

  // ---- layer 1: transform-then-aggregate -----------------------------------
  // Ys = x_sent @ [W0 | W2 | loop]  (N=768), Yd = x_doc @ [W1 | loop] (N=512)
  gemm_bf16_wmma<float><<<dim3(NS / 16, 6), 32, 0, stream>>>(x_sent, 768, WTs1, Ys, 768, 768);
  gemm_bf16_wmma<float><<<dim3(ND / 16, 4), 32, 0, stream>>>(x_doc, 768, WTd1, Yd, 512, 768);

  hipMemsetAsync(aggA, 0, (size_t)NS * 256 * 4, stream);
  hipMemsetAsync(aggB, 0, (size_t)NS * 256 * 4, stream);
  hipMemsetAsync(aggD, 0, (size_t)ND * 256 * 4, stream);
  {
    long t;
    t = (long)Esd * 64; scatter_add<<<(t + 255) / 256, 256, 0, stream>>>(Ys, 768, 0,   src_sd, dst_sd, Esd, aggD, 256);
    t = (long)Ess * 64; scatter_add<<<(t + 255) / 256, 256, 0, stream>>>(Ys, 768, 256, src_ss, dst_ss, Ess, aggB, 256);
    t = (long)Eds * 64; scatter_add<<<(t + 255) / 256, 256, 0, stream>>>(Yd, 512, 0,   src_ds, dst_ds, Eds, aggA, 256);
  }
  combine1<<<NS, 256, 0, stream>>>(aggA, degDs, aggB, degSs, Ys, 768, 512, bias1, h1s, 256, NS);
  combine1<<<ND, 256, 0, stream>>>(aggD, degSd, (const float*)nullptr, (const float*)nullptr,
                                   Yd, 512, 256, bias1, h1d, 256, ND);

  // ---- layer 2 -------------------------------------------------------------
  gemm_bf16_wmma<__bf16><<<dim3(NS / 16, 3), 32, 0, stream>>>(h1s, 256, WTs2, Ys, 384, 256);
  gemm_bf16_wmma<__bf16><<<dim3(ND / 16, 2), 32, 0, stream>>>(h1d, 256, WTd2, Yd, 256, 256);

  hipMemsetAsync(aggA, 0, (size_t)NS * 128 * 4, stream);
  hipMemsetAsync(aggB, 0, (size_t)NS * 128 * 4, stream);
  hipMemsetAsync(aggD, 0, (size_t)ND * 128 * 4, stream);
  {
    long t;
    t = (long)Esd * 32; scatter_add<<<(t + 255) / 256, 256, 0, stream>>>(Ys, 384, 0,   src_sd, dst_sd, Esd, aggD, 128);
    t = (long)Ess * 32; scatter_add<<<(t + 255) / 256, 256, 0, stream>>>(Ys, 384, 128, src_ss, dst_ss, Ess, aggB, 128);
    t = (long)Eds * 32; scatter_add<<<(t + 255) / 256, 256, 0, stream>>>(Yd, 256, 0,   src_ds, dst_ds, Eds, aggA, 128);
  }

  // ---- fused combine + readout --------------------------------------------
  out_init<<<1, 64, 0, stream>>>(out, b_score, G);
  combine2_readout<<<(NS + 7) / 8, 256, 0, stream>>>(aggA, degDs, aggB, degSs, Ys, 384, 256,
                                                     bias2, w_score, gid_sent, out, NS);
  combine2_readout<<<(ND + 7) / 8, 256, 0, stream>>>(aggD, degSd, (const float*)nullptr, (const float*)nullptr,
                                                     Yd, 256, 128, bias2, w_score, gid_doc, out, ND);
}
